// Encoder_10754598109893
// MI455X (gfx1250) — matrix operand order repaired and checked by reading, not compiled, not run on hardware
//
#include <hip/hip_runtime.h>
#include <hip/hip_bf16.h>
#include <math.h>

// ---------------------------------------------------------------------------
// 2-layer bidirectional GRU (B=256, T=256, D=64, H=256) for gfx1250 (MI455X).
// Batch rows are independent recurrent chains -> 16 WGs x 16 rows per
// direction, persistent scan over T with only workgroup barriers.
// Recurrent + input GEMMs fused into v_wmma_f32_16x16x32_bf16 K-loops over
// the concatenated [x_t | h] operand vs packed [W_ih | W_hh] weights (L2-hot).
// 16 waves/WG, 3 N-tiles/wave: accumulator footprint 48 VGPR/lane -> no
// spills; x-part vs h-part handled by two separate rolled K-loops (no branch
// anywhere near WMMA; EXEC stays all-1s).
// ---------------------------------------------------------------------------

typedef __bf16 bf16_t;
typedef __attribute__((ext_vector_type(16))) __bf16 v16bf;
typedef __attribute__((ext_vector_type(8)))  __bf16 v8bf;
typedef __attribute__((ext_vector_type(8)))  float  v8f;

static __device__ __forceinline__ v8bf ld8(const bf16_t* p) {
    return *reinterpret_cast<const v8bf*>(p);
}
static __device__ __forceinline__ v16bf cat16(v8bf lo, v8bf hi) {
    v16bf r;
#pragma unroll
    for (int i = 0; i < 8; ++i) { r[i] = lo[i]; r[8 + i] = hi[i]; }
    return r;
}

// ------------------------------- packing -----------------------------------

__global__ void cvt_f32_to_bf16(const float* __restrict__ src,
                                bf16_t* __restrict__ dst, int n) {
    for (int i = blockIdx.x * blockDim.x + threadIdx.x; i < n;
         i += gridDim.x * blockDim.x)
        dst[i] = (bf16_t)src[i];
}

// Wc[dir][n][k]: k < C -> w_ih[dir][n][k] ; else w_hh[dir][n][k-C]   (bf16)
__global__ void pack_weights(const float* __restrict__ w_ih,
                             const float* __restrict__ w_hh,
                             bf16_t* __restrict__ Wc, int C) {
    const int AK = C + 256;
    const int total = 2 * 768 * AK;
    for (int i = blockIdx.x * blockDim.x + threadIdx.x; i < total;
         i += gridDim.x * blockDim.x) {
        int k = i % AK;
        int n = (i / AK) % 768;
        int dir = i / (AK * 768);
        float v = (k < C) ? w_ih[(dir * 768 + n) * C + k]
                          : w_hh[(dir * 768 + n) * 256 + (k - C)];
        Wc[i] = (bf16_t)v;
    }
}

// ------------------------------ GRU layer ----------------------------------
// grid = 32 WGs: blockIdx.x>>4 = direction, (blockIdx.x&15)*16 = batch row b0.
// block = 512 threads = 16 waves (wave32). Each wave owns 3 N-tiles (48 of the
// 768 gate columns); each A fragment (ds_load_b128 pair) feeds 3 WMMAs.

template <int C, bool TMAJOR, bool PERSTEP>
__global__ __launch_bounds__(512)
void gru_layer_kernel(const bf16_t* __restrict__ X,   // TMAJOR ? [T][B][C] : [B][T][C]
                      const bf16_t* __restrict__ W,   // [2][768][C+256]
                      const float*  __restrict__ b_ih,// [2][768]
                      const float*  __restrict__ b_hh,// [2][768]
                      bf16_t* __restrict__ Yseq,      // PERSTEP: [T][B][512]
                      float*  __restrict__ Yfin)      // !PERSTEP: [B][512]
{
    constexpr int H = 256, G = 768, Tn = 256, Bn = 256, AK = C + H;
    const int tid  = threadIdx.x;
    const int wave = __builtin_amdgcn_readfirstlane(tid >> 5);  // scalar wave id
    const int lane = tid & 31;
    const int sub  = lane & 15;     // row (A) / column (B) within tile
    const int hl   = lane >> 4;     // lane half
    const int dir  = blockIdx.x >> 4;
    const int b0   = (blockIdx.x & 15) * 16;

    extern __shared__ char smem[];
    bf16_t* Asm  = (bf16_t*)smem;                                     // 16 x AK bf16
    float*  ghs  = (float*)(smem + (size_t)16 * AK * sizeof(bf16_t)); // 16 x 768
    float*  gis  = ghs  + 16 * G;                                     // 16 x 256
    float*  hfp  = gis  + 16 * H;                                     // 16 x 256
    float*  bihs = hfp  + 16 * H;                                     // 768
    float*  bhhs = bihs + G;                                          // 768

    const bf16_t* Wd = W + (size_t)dir * G * AK;
    // base of this wave's 3 weight tiles; lane sub picks the gate column
    const bf16_t* Wtile = Wd + (size_t)(wave * 48 + sub) * AK;

    for (int i = tid; i < G; i += blockDim.x) {
        bihs[i] = b_ih[dir * G + i];
        bhhs[i] = b_hh[dir * G + i];
    }
    for (int i = tid; i < 16 * H; i += blockDim.x) hfp[i] = 0.f;
    for (int i = tid; i < 16 * AK; i += blockDim.x) Asm[i] = (bf16_t)0.f;

    // stage x_{t(s=0)} tile (16 rows x C) into Asm[:, 0..C)
    {
        const int t0 = dir ? (Tn - 1) : 0;
        for (int e = tid * 8; e < 16 * C; e += blockDim.x * 8) {
            int m = e / C, d = e % C;
            size_t row = TMAJOR ? ((size_t)t0 * Bn + (b0 + m))
                                : ((size_t)(b0 + m) * Tn + t0);
            *reinterpret_cast<v8bf*>(&Asm[m * AK + d]) =
                *reinterpret_cast<const v8bf*>(&X[row * C + d]);
        }
    }
    __syncthreads();

    for (int s = 0; s < Tn; ++s) {
        const int t = dir ? (Tn - 1 - s) : s;

        // ---- fused GEMM: x-part then h-part, branchless around WMMA ------
        v8f accx[3], acch[3];
#pragma unroll
        for (int i = 0; i < 3; ++i) {
            v8f zz = {};
            accx[i] = zz;
            acch[i] = zz;
            __builtin_prefetch(Wtile + (size_t)i * 16 * AK, 0, 1); // global_prefetch_b8
        }
#pragma unroll 2
        for (int kb = 0; kb < C; kb += 32) {          // gi: x_t x W_ih
            v16bf a = cat16(ld8(&Asm[sub * AK + kb + hl * 8]),
                            ld8(&Asm[sub * AK + kb + hl * 8 + 16]));
#pragma unroll
            for (int i = 0; i < 3; ++i) {
                const bf16_t* Wrow = Wtile + (size_t)i * 16 * AK;
                v16bf b = cat16(ld8(&Wrow[kb + hl * 8]),
                                ld8(&Wrow[kb + hl * 8 + 16]));
                accx[i] = __builtin_amdgcn_wmma_f32_16x16x32_bf16(
                    false, a, false, b, (short)0, accx[i], false, false);
            }
        }
#pragma unroll 2
        for (int kb = C; kb < AK; kb += 32) {         // gh: h x W_hh
            v16bf a = cat16(ld8(&Asm[sub * AK + kb + hl * 8]),
                            ld8(&Asm[sub * AK + kb + hl * 8 + 16]));
#pragma unroll
            for (int i = 0; i < 3; ++i) {
                const bf16_t* Wrow = Wtile + (size_t)i * 16 * AK;
                v16bf b = cat16(ld8(&Wrow[kb + hl * 8]),
                                ld8(&Wrow[kb + hl * 8 + 16]));
                acch[i] = __builtin_amdgcn_wmma_f32_16x16x32_bf16(
                    false, a, false, b, (short)0, acch[i], false, false);
            }
        }
        // store pre-activations to LDS (LDS stores are EXEC-safe under any mask)
#pragma unroll
        for (int i = 0; i < 3; ++i) {
            const int nb = wave * 48 + i * 16;
            if (nb < 512) {   // r,z gates: x-part + h-part merged
#pragma unroll
                for (int v = 0; v < 8; ++v)
                    ghs[(v + hl * 8) * G + nb + sub] = accx[i][v] + acch[i][v];
            } else {          // n gate: keep x-part (gi) and h-part (gh) split
#pragma unroll
                for (int v = 0; v < 8; ++v) {
                    gis[(v + hl * 8) * H + (nb - 512) + sub] = accx[i][v];
                    ghs[(v + hl * 8) * G + nb + sub]         = acch[i][v];
                }
            }
        }
        __syncthreads();   // GEMM done: ghs/gis ready, Asm free to modify

        // ---- gate math + state update; overlap staging of next x tile ----
        {
            const int m  = tid >> 5;   // row 0..15
            const int c0 = tid & 31;
#pragma unroll
            for (int j = 0; j < 8; ++j) {
                const int c = c0 + j * 32;
                float pre_r = ghs[m * G + c]       + bihs[c]       + bhhs[c];
                float pre_z = ghs[m * G + 256 + c] + bihs[256 + c] + bhhs[256 + c];
                float gh_n  = ghs[m * G + 512 + c] + bhhs[512 + c];
                float gi_n  = gis[m * H + c]       + bihs[512 + c];
                float r = 1.f / (1.f + __expf(-pre_r));
                float z = 1.f / (1.f + __expf(-pre_z));
                float n = tanhf(gi_n + r * gh_n);
                float hp = hfp[m * H + c];
                float hn = (1.f - z) * n + z * hp;
                hfp[m * H + c]       = hn;
                Asm[m * AK + C + c]  = (bf16_t)hn;   // bf16 A operand, next step
                if (PERSTEP)
                    Yseq[((size_t)t * Bn + (b0 + m)) * 512 + dir * H + c] = (bf16_t)hn;
            }
        }
        if (s + 1 < Tn) {   // stage x_{t(s+1)} while gate math retires
            const int tn = dir ? (Tn - 2 - s) : (s + 1);
            for (int e = tid * 8; e < 16 * C; e += blockDim.x * 8) {
                int m = e / C, d = e % C;
                size_t row = TMAJOR ? ((size_t)tn * Bn + (b0 + m))
                                    : ((size_t)(b0 + m) * Tn + tn);
                *reinterpret_cast<v8bf*>(&Asm[m * AK + d]) =
                    *reinterpret_cast<const v8bf*>(&X[row * C + d]);
            }
        }
        __syncthreads();   // Asm (x + h) ready for next GEMM
    }

    if (!PERSTEP) {
        const int m  = tid >> 5;
        const int c0 = tid & 31;
#pragma unroll
        for (int j = 0; j < 8; ++j) {
            const int c = c0 + j * 32;
            Yfin[(size_t)(b0 + m) * 512 + dir * H + c] = hfp[m * H + c];
        }
    }
}

// ------------------------------ launcher -----------------------------------

static inline size_t align256(size_t x) { return (x + 255) & ~(size_t)255; }

extern "C" void kernel_launch(void* const* d_in, const int* in_sizes, int n_in,
                              void* d_out, int out_size, void* d_ws, size_t ws_size,
                              hipStream_t stream) {
    (void)in_sizes; (void)n_in; (void)out_size; (void)ws_size;
    constexpr int B = 256, T = 256, D = 64, H = 256;

    const float* x     = (const float*)d_in[0];
    const float* w_ih0 = (const float*)d_in[1];
    const float* w_hh0 = (const float*)d_in[2];
    const float* b_ih0 = (const float*)d_in[3];
    const float* b_hh0 = (const float*)d_in[4];
    const float* w_ih1 = (const float*)d_in[5];
    const float* w_hh1 = (const float*)d_in[6];
    const float* b_ih1 = (const float*)d_in[7];
    const float* b_hh1 = (const float*)d_in[8];

    char* ws = (char*)d_ws;
    size_t off = 0;
    bf16_t* xbf = (bf16_t*)(ws + off); off += align256((size_t)B * T * D * 2);      // 8 MB
    bf16_t* Wc0 = (bf16_t*)(ws + off); off += align256((size_t)2 * 768 * (D + H) * 2);
    bf16_t* Wc1 = (bf16_t*)(ws + off); off += align256((size_t)2 * 768 * (2 * H + H) * 2);
    bf16_t* h1  = (bf16_t*)(ws + off); off += (size_t)T * B * 512 * 2;              // 64 MB

    // 1) pack inputs / weights to bf16 (weights stay L2-resident, ~3.3 MB)
    cvt_f32_to_bf16<<<(B * T * D + 255) / 256, 256, 0, stream>>>(x, xbf, B * T * D);
    pack_weights<<<1024, 256, 0, stream>>>(w_ih0, w_hh0, Wc0, D);
    pack_weights<<<2048, 256, 0, stream>>>(w_ih1, w_hh1, Wc1, 2 * H);

    // LDS: 16*AK*2 (A) + 16*768*4 (gh) + 16*256*4 (gi_n) + 16*256*4 (h) + 2*768*4 (bias)
    const size_t shm0 = (size_t)16 * (D + H) * 2     + 16 * 768 * 4 + 16 * 256 * 4
                      + 16 * 256 * 4 + 2 * 768 * 4;  // ~96 KB
    const size_t shm1 = (size_t)16 * (2 * H + H) * 2 + 16 * 768 * 4 + 16 * 256 * 4
                      + 16 * 256 * 4 + 2 * 768 * 4;  // ~110 KB

    // 2) layer 0: x (B-major) -> h1 [T][B][512] bf16, both directions
    gru_layer_kernel<D, false, true><<<32, 512, shm0, stream>>>(
        xbf, Wc0, b_ih0, b_hh0, h1, nullptr);

    // 3) layer 1: h1 (T-major) -> final fwd(t=T-1) | bwd(t=0) -> d_out fp32
    gru_layer_kernel<2 * H, true, false><<<32, 512, shm1, stream>>>(
        h1, Wc1, b_ih1, b_hh1, nullptr, (float*)d_out);
}
